// ASAP_Pooling_43310450213008
// MI455X (gfx1250) — matrix operand order repaired and checked by reading, not compiled, not run on hardware
//
#include <hip/hip_runtime.h>
#include <math.h>

// ---------------------------------------------------------------------------
// CDNA5 (gfx1250) implementation of the EGNN + ASAP pooling forward pass.
// wave32; WMMA bf16 16x16x32 for the edge MLP / node update / M_q GEMMs.
// ---------------------------------------------------------------------------

typedef __attribute__((ext_vector_type(16))) __bf16 v16bf;
typedef __attribute__((ext_vector_type(8)))  __bf16 v8bf;
typedef __attribute__((ext_vector_type(8)))  float  v8f;

union V16 { v16bf v; v8bf h[2]; };

// Intra-wave LDS phase fence: DS ops are in-order per wave; this makes the
// compiler keep the ordering and drains DScnt (CDNA5 split counters).
#define LDS_FENCE() asm volatile("s_wait_dscnt 0" ::: "memory")

__device__ __forceinline__ float sigmoidf_(float x) { return 1.0f / (1.0f + __expf(-x)); }
__device__ __forceinline__ float siluf_(float x)    { return x * sigmoidf_(x); }

__device__ __forceinline__ void edge_nodes(int e, int E, const int* __restrict__ src,
                                           const int* __restrict__ dst, int& i0, int& i1) {
  if (e < E) { i0 = src[e]; i1 = dst[e]; } else { i0 = e - E; i1 = i0; }
}

__device__ __forceinline__ void atomicMaxF(float* p, float v) {
  __hip_atomic_fetch_max(p, v, __ATOMIC_RELAXED, __HIP_MEMORY_SCOPE_AGENT);
}

static constexpr int C = 96;     // channels / hidden
static constexpr int AS = 200;   // LDS A-row stride (bf16 elems), 16B aligned, conflict-padded
static constexpr int ASQ = 104;  // LDS A-row stride for K=96 staging

// ---------------------------------------------------------------------------
// init: counts=1 (self loop), stat accumulators=0, agg=0, X_q=-inf, smax=-inf
// ---------------------------------------------------------------------------
__global__ void asap_init_kernel(int* cnt0, int* cnt1, float* agg, float* Xq,
                                 float* smax, float* den, float* aggr,
                                 float* sAcc, int N) {
  int i = blockIdx.x * blockDim.x + threadIdx.x;
  if (i < 8) sAcc[i] = 0.0f;
  if (i < N) {
    cnt0[i] = 1; cnt1[i] = 1;
    smax[i] = -__builtin_inff();
    den[i]  = 0.0f;
    aggr[i] = 0.0f;
  }
  if (i < N * C) {
    agg[i] = 0.0f;
    Xq[i]  = -__builtin_inff();
  }
}

__global__ void asap_zero_kernel(float* p, int n) {
  int i = blockIdx.x * blockDim.x + threadIdx.x;
  if (i < n) p[i] = 0.0f;
}

// ---------------------------------------------------------------------------
// per-node gather multiplicity counts (real edges; self loops pre-counted)
// ---------------------------------------------------------------------------
__global__ void asap_count_kernel(const int* __restrict__ src, const int* __restrict__ dst,
                                  int* cnt0, int* cnt1, int E) {
  int e = blockIdx.x * blockDim.x + threadIdx.x;
  if (e < E) {
    atomicAdd(&cnt0[src[e]], 1);
    atomicAdd(&cnt1[dst[e]], 1);
  }
}

// ---------------------------------------------------------------------------
// gathered-LayerNorm moments: sum / sumsq of x weighted by gather counts
// ---------------------------------------------------------------------------
__global__ void asap_moments_kernel(const float* __restrict__ x,
                                    const int* __restrict__ cnt0,
                                    const int* __restrict__ cnt1,
                                    float* sAcc, int N) {
  int n = blockIdx.x * blockDim.x + threadIdx.x;
  float s = 0.0f, q = 0.0f, c0 = 0.0f, c1 = 0.0f;
  if (n < N) {
    const float* row = x + (size_t)n * C;
    #pragma unroll 8
    for (int c = 0; c < C; ++c) { float v = row[c]; s += v; q += v * v; }
    c0 = (float)cnt0[n]; c1 = (float)cnt1[n];
  }
  float v0 = c0 * s, v1 = c0 * q, v2 = c1 * s, v3 = c1 * q;
  #pragma unroll
  for (int off = 16; off > 0; off >>= 1) {
    v0 += __shfl_down(v0, off, 32);
    v1 += __shfl_down(v1, off, 32);
    v2 += __shfl_down(v2, off, 32);
    v3 += __shfl_down(v3, off, 32);
  }
  if ((threadIdx.x & 31) == 0) {
    unsafeAtomicAdd(&sAcc[0], v0);
    unsafeAtomicAdd(&sAcc[1], v1);
    unsafeAtomicAdd(&sAcc[2], v2);
    unsafeAtomicAdd(&sAcc[3], v3);
  }
}

__global__ void asap_finalize_stats_kernel(const float* sAcc, float* sFin, int Ef) {
  if (threadIdx.x == 0 && blockIdx.x == 0) {
    float inv = 1.0f / ((float)Ef * (float)C);
    float mu0 = sAcc[0] * inv, var0 = sAcc[1] * inv - mu0 * mu0;
    float mu1 = sAcc[2] * inv, var1 = sAcc[3] * inv - mu1 * mu1;
    sFin[0] = mu0; sFin[1] = rsqrtf(var0 + 1e-5f);
    sFin[2] = mu1; sFin[3] = rsqrtf(var1 + 1e-5f);
  }
}

// ---------------------------------------------------------------------------
// weight prep: bf16, n-major (so WMMA B fragments are contiguous 32B loads)
// ---------------------------------------------------------------------------
__global__ void asap_prep_weights_kernel(const float* __restrict__ mw1, const float* __restrict__ mw2,
                                         const float* __restrict__ uw1, const float* __restrict__ uw2,
                                         const float* __restrict__ lqw,
                                         __bf16* w1T, float* w1last, __bf16* w2T,
                                         __bf16* u1T, __bf16* u2T, __bf16* lqT) {
  int idx = blockIdx.x * blockDim.x + threadIdx.x;
  if (idx >= C * 192) return;
  int n = idx / 192, k = idx % 192;
  w1T[n * 192 + k] = (__bf16)mw1[k * C + n];
  u1T[n * 192 + k] = (__bf16)uw1[k * C + n];
  if (k < C) {
    w2T[n * C + k] = (__bf16)mw2[k * C + n];
    u2T[n * C + k] = (__bf16)uw2[k * C + n];
    lqT[n * C + k] = (__bf16)lqw[k * C + n];
  }
  if (k == 0) w1last[n] = mw1[192 * C + n];
}

// ---------------------------------------------------------------------------
// A-fragment loader from a linear bf16 LDS row block (k order shared with load_b_frag):
// lane = hi*16+nlo holds row m=nlo, k = hi*16+{0..15} (contiguous).
// ---------------------------------------------------------------------------
__device__ __forceinline__ V16 load_a_frag(const __bf16* rows, int stride, int nlo, int hi, int t) {
  V16 f;
  const __bf16* p = rows + nlo * stride + t * 32 + hi * 16;
  f.h[0] = *(const v8bf*)p;
  f.h[1] = *(const v8bf*)(p + 8);
  return f;
}

// B fragment: element e of lane (hi,nlo) is (k = t*32 + hi*16 + e, n = nlo):
// contiguous 16 bf16 in n-major weight storage.
__device__ __forceinline__ v16bf load_b_frag(const __bf16* __restrict__ wT, int kstride,
                                             int nlo, int hi, int nt, int t) {
  return *(const v16bf*)(wT + (nt * 16 + nlo) * kstride + t * 32 + hi * 16);
}

// ---------------------------------------------------------------------------
// Edge MLP: per 16-edge tile per wave.
//   minp=[gln(x[i1]), gln(x[i0]), dist] -> silu(W1) -> silu(W2) -> sigmoid gate
//   scatter gate*h into agg[i1] with float atomics.
// ---------------------------------------------------------------------------
__global__ __launch_bounds__(64) void asap_edge_mlp_kernel(
    const float* __restrict__ x, const float* __restrict__ pos,
    const int* __restrict__ src, const int* __restrict__ dst,
    const float* __restrict__ sFin,
    const float* __restrict__ lnw, const float* __restrict__ lnb,
    const __bf16* __restrict__ w1T, const float* __restrict__ w1last,
    const float* __restrict__ b1,
    const __bf16* __restrict__ w2T, const float* __restrict__ b2,
    const float* __restrict__ attw, const float* __restrict__ attb,
    float* __restrict__ agg, int E, int N) {
  __shared__ __bf16 Abuf[2][16 * AS];
  __shared__ __bf16 h1buf[2][16 * C];
  __shared__ float  h2buf[2][16 * C];
  __shared__ float  distb[2][16];
  __shared__ float  gateb[2][16];
  __shared__ int    idxb[2][16];
  __shared__ float  lnw_s[C], lnb_s[C], attw_s[C];

  const int lane = threadIdx.x & 31;
  const int wid  = threadIdx.x >> 5;
  const int nlo  = lane & 15;
  const int hi   = lane >> 4;

  for (int i = threadIdx.x; i < C; i += 64) {
    lnw_s[i] = lnw[i]; lnb_s[i] = lnb[i]; attw_s[i] = attw[i];
  }
  __syncthreads();

  __builtin_prefetch(w1T + threadIdx.x * 64, 0, 1);  // warm L2 with weights
  __builtin_prefetch(w2T + threadIdx.x * 32, 0, 1);

  const int Ef = E + N;
  const int NT = (Ef + 15) >> 4;
  const float mu0 = sFin[0], rs0 = sFin[1], mu1 = sFin[2], rs1 = sFin[3];
  const float attbv = attb[0];

  __bf16* Ab  = Abuf[wid];
  __bf16* h1  = h1buf[wid];
  float*  h2  = h2buf[wid];
  float*  db  = distb[wid];
  float*  gb  = gateb[wid];
  int*    ib  = idxb[wid];

  const int gw = blockIdx.x * 2 + wid;
  const int tw = gridDim.x * 2;

  for (int tile = gw; tile < NT; tile += tw) {
    // ---- stage gln(xi)|gln(xj) rows + dist into LDS (coalesced) ----
    for (int m = 0; m < 16; ++m) {
      int e = tile * 16 + m;
      bool ok = e < Ef;
      int ec = ok ? e : 0;
      int i0, i1; edge_nodes(ec, E, src, dst, i0, i1);
      #pragma unroll
      for (int j = 0; j < 3; ++j) {
        int c = lane + 32 * j;
        float vi = x[(size_t)i1 * C + c];
        float vj = x[(size_t)i0 * C + c];
        Ab[m * AS + c]     = (__bf16)(((vi - mu1) * rs1) * lnw_s[c] + lnb_s[c]);
        Ab[m * AS + C + c] = (__bf16)(((vj - mu0) * rs0) * lnw_s[c] + lnb_s[c]);
      }
      if (lane == 0) {
        float dx = pos[i1 * 3 + 0] - pos[i0 * 3 + 0];
        float dy = pos[i1 * 3 + 1] - pos[i0 * 3 + 1];
        float dz = pos[i1 * 3 + 2] - pos[i0 * 3 + 2];
        db[m] = sqrtf(dx * dx + dy * dy + dz * dz);
        ib[m] = ok ? i1 : -1;
      }
    }
    LDS_FENCE();

    // ---- cache A fragments (K=192 -> 6 k-steps) + dist per M ----
    V16 af[6];
    #pragma unroll
    for (int t = 0; t < 6; ++t) af[t] = load_a_frag(Ab, AS, nlo, hi, t);
    float dv[8];
    #pragma unroll
    for (int r = 0; r < 8; ++r) dv[r] = db[r + hi * 8];

    // ---- matmul1 + dist rank-1 + bias + silu -> h1 (bf16) ----
    #pragma unroll
    for (int nt = 0; nt < 6; ++nt) {
      int n = nt * 16 + nlo;
      v8f acc = {0.f, 0.f, 0.f, 0.f, 0.f, 0.f, 0.f, 0.f};
      #pragma unroll
      for (int t = 0; t < 6; ++t) {
        v16bf bv = load_b_frag(w1T, 192, nlo, hi, nt, t);
        acc = __builtin_amdgcn_wmma_f32_16x16x32_bf16(false, af[t].v, false, bv,
                                                      (short)0, acc, false, false);
      }
      float wl = w1last[n], bb = b1[n];
      #pragma unroll
      for (int r = 0; r < 8; ++r) {
        float h = acc[r] + dv[r] * wl + bb;
        h1[(r + hi * 8) * C + n] = (__bf16)siluf_(h);
      }
    }
    LDS_FENCE();

    // ---- matmul2 (K=96 -> 3 k-steps) + bias + silu -> h2 (f32) ----
    V16 hf[3];
    #pragma unroll
    for (int t = 0; t < 3; ++t) hf[t] = load_a_frag(h1, C, nlo, hi, t);
    #pragma unroll
    for (int nt = 0; nt < 6; ++nt) {
      int n = nt * 16 + nlo;
      v8f acc = {0.f, 0.f, 0.f, 0.f, 0.f, 0.f, 0.f, 0.f};
      #pragma unroll
      for (int t = 0; t < 3; ++t) {
        v16bf bv = load_b_frag(w2T, C, nlo, hi, nt, t);
        acc = __builtin_amdgcn_wmma_f32_16x16x32_bf16(false, hf[t].v, false, bv,
                                                      (short)0, acc, false, false);
      }
      float bb = b2[n];
      #pragma unroll
      for (int r = 0; r < 8; ++r)
        h2[(r + hi * 8) * C + n] = siluf_(acc[r] + bb);
    }
    LDS_FENCE();

    // ---- attention gate: sigmoid(h2 . att_w + att_b) per edge ----
    float part = 0.0f;
    #pragma unroll 8
    for (int j = 0; j < 48; ++j)
      part += h2[nlo * C + hi * 48 + j] * attw_s[hi * 48 + j];
    float other = __shfl_xor(part, 16, 32);
    float g = sigmoidf_(part + other + attbv);
    if (hi == 0) gb[nlo] = g;
    LDS_FENCE();

    // ---- scatter gate*h2 into agg[i1] ----
    for (int m = 0; m < 16; ++m) {
      int i1 = ib[m];
      if (i1 >= 0) {
        float gm = gb[m];
        #pragma unroll
        for (int j = 0; j < 3; ++j) {
          int c = lane + 32 * j;
          unsafeAtomicAdd(&agg[(size_t)i1 * C + c], gm * h2[m * C + c]);
        }
      }
    }
  }
}

// ---------------------------------------------------------------------------
// Node update: x_pool = silu([x,agg] @ u1 + b1) @ u2 + b2 + x   (WMMA)
// ---------------------------------------------------------------------------
__global__ __launch_bounds__(64) void asap_node_update_kernel(
    const float* __restrict__ x, const float* __restrict__ agg,
    const __bf16* __restrict__ u1T, const float* __restrict__ u1b,
    const __bf16* __restrict__ u2T, const float* __restrict__ u2b,
    float* __restrict__ x_pool, int N) {
  __shared__ __bf16 Abuf[2][16 * AS];
  __shared__ __bf16 h1buf[2][16 * C];

  const int lane = threadIdx.x & 31;
  const int wid  = threadIdx.x >> 5;
  const int nlo  = lane & 15;
  const int hi   = lane >> 4;
  __bf16* Ab = Abuf[wid];
  __bf16* h1 = h1buf[wid];

  const int NT = (N + 15) >> 4;
  const int gw = blockIdx.x * 2 + wid;
  const int tw = gridDim.x * 2;

  for (int tile = gw; tile < NT; tile += tw) {
    for (int m = 0; m < 16; ++m) {
      int node = tile * 16 + m;
      int nc = node < N ? node : N - 1;
      #pragma unroll
      for (int j = 0; j < 3; ++j) {
        int c = lane + 32 * j;
        Ab[m * AS + c]     = (__bf16)x[(size_t)nc * C + c];
        Ab[m * AS + C + c] = (__bf16)agg[(size_t)nc * C + c];
      }
    }
    LDS_FENCE();

    V16 af[6];
    #pragma unroll
    for (int t = 0; t < 6; ++t) af[t] = load_a_frag(Ab, AS, nlo, hi, t);

    #pragma unroll
    for (int nt = 0; nt < 6; ++nt) {
      int n = nt * 16 + nlo;
      v8f acc = {0.f, 0.f, 0.f, 0.f, 0.f, 0.f, 0.f, 0.f};
      #pragma unroll
      for (int t = 0; t < 6; ++t) {
        v16bf bv = load_b_frag(u1T, 192, nlo, hi, nt, t);
        acc = __builtin_amdgcn_wmma_f32_16x16x32_bf16(false, af[t].v, false, bv,
                                                      (short)0, acc, false, false);
      }
      float bb = u1b[n];
      #pragma unroll
      for (int r = 0; r < 8; ++r)
        h1[(r + hi * 8) * C + n] = (__bf16)siluf_(acc[r] + bb);
    }
    LDS_FENCE();

    V16 hf[3];
    #pragma unroll
    for (int t = 0; t < 3; ++t) hf[t] = load_a_frag(h1, C, nlo, hi, t);
    #pragma unroll
    for (int nt = 0; nt < 6; ++nt) {
      int n = nt * 16 + nlo;
      v8f acc = {0.f, 0.f, 0.f, 0.f, 0.f, 0.f, 0.f, 0.f};
      #pragma unroll
      for (int t = 0; t < 3; ++t) {
        v16bf bv = load_b_frag(u2T, C, nlo, hi, nt, t);
        acc = __builtin_amdgcn_wmma_f32_16x16x32_bf16(false, hf[t].v, false, bv,
                                                      (short)0, acc, false, false);
      }
      float bb = u2b[n];
      #pragma unroll
      for (int r = 0; r < 8; ++r) {
        int node = tile * 16 + r + hi * 8;
        if (node < N)
          x_pool[(size_t)node * C + n] = acc[r] + bb + x[(size_t)node * C + n];
      }
    }
  }
}

// ---------------------------------------------------------------------------
// X_q = scatter-max over i0 of x_pool[i1]   (float atomic max)
// ---------------------------------------------------------------------------
__global__ void asap_xqmax_kernel(const float* __restrict__ x_pool,
                                  const int* __restrict__ src, const int* __restrict__ dst,
                                  float* Xq, int E, int N) {
  int gid = blockIdx.x * blockDim.x + threadIdx.x;
  int e = gid >> 5, lane = gid & 31;
  int Ef = E + N;
  if (e >= Ef) return;
  int i0, i1; edge_nodes(e, E, src, dst, i0, i1);
  #pragma unroll
  for (int j = 0; j < 3; ++j) {
    int c = lane + 32 * j;
    atomicMaxF(&Xq[(size_t)i0 * C + c], x_pool[(size_t)i1 * C + c]);
  }
}

// ---------------------------------------------------------------------------
// M_q = X_q @ linq_w + linq_b  (WMMA, K=96)
// ---------------------------------------------------------------------------
__global__ __launch_bounds__(64) void asap_mq_kernel(
    const float* __restrict__ Xq, const __bf16* __restrict__ lqT,
    const float* __restrict__ lqb, float* __restrict__ Mq, int N) {
  __shared__ __bf16 Abuf[2][16 * ASQ];
  const int lane = threadIdx.x & 31;
  const int wid  = threadIdx.x >> 5;
  const int nlo  = lane & 15;
  const int hi   = lane >> 4;
  __bf16* Ab = Abuf[wid];

  const int NT = (N + 15) >> 4;
  const int gw = blockIdx.x * 2 + wid;
  const int tw = gridDim.x * 2;

  for (int tile = gw; tile < NT; tile += tw) {
    for (int m = 0; m < 16; ++m) {
      int node = tile * 16 + m;
      int nc = node < N ? node : N - 1;
      #pragma unroll
      for (int j = 0; j < 3; ++j) {
        int c = lane + 32 * j;
        Ab[m * ASQ + c] = (__bf16)Xq[(size_t)nc * C + c];
      }
    }
    LDS_FENCE();

    V16 af[3];
    #pragma unroll
    for (int t = 0; t < 3; ++t) af[t] = load_a_frag(Ab, ASQ, nlo, hi, t);

    #pragma unroll
    for (int nt = 0; nt < 6; ++nt) {
      int n = nt * 16 + nlo;
      v8f acc = {0.f, 0.f, 0.f, 0.f, 0.f, 0.f, 0.f, 0.f};
      #pragma unroll
      for (int t = 0; t < 3; ++t) {
        v16bf bv = load_b_frag(lqT, C, nlo, hi, nt, t);
        acc = __builtin_amdgcn_wmma_f32_16x16x32_bf16(false, af[t].v, false, bv,
                                                      (short)0, acc, false, false);
      }
      float bb = lqb[n];
      #pragma unroll
      for (int r = 0; r < 8; ++r) {
        int node = tile * 16 + r + hi * 8;
        if (node < N) Mq[(size_t)node * C + n] = acc[r] + bb;
      }
    }
    LDS_FENCE();
  }
}

// ---------------------------------------------------------------------------
// per-node halves of the GAT score: a = Mq . gat_w[:96], b = x_pool . gat_w[96:]
// ---------------------------------------------------------------------------
__global__ void asap_node_dots_kernel(const float* __restrict__ Mq,
                                      const float* __restrict__ x_pool,
                                      const float* __restrict__ gatw,
                                      float* a_node, float* b_node, int N) {
  int n = blockIdx.x * blockDim.x + threadIdx.x;
  if (n >= N) return;
  float a = 0.0f, b = 0.0f;
  #pragma unroll 8
  for (int c = 0; c < C; ++c) {
    a += Mq[(size_t)n * C + c] * gatw[c];
    b += x_pool[(size_t)n * C + c] * gatw[C + c];
  }
  a_node[n] = a; b_node[n] = b;
}

// score passes: leaky-relu score + stable scatter-softmax over i0
__global__ void asap_score1_kernel(const int* __restrict__ src, const int* __restrict__ dst,
                                   const float* __restrict__ a_node, const float* __restrict__ b_node,
                                   const float* __restrict__ gatb,
                                   float* s_e, float* smax, int E, int N) {
  int e = blockIdx.x * blockDim.x + threadIdx.x;
  int Ef = E + N;
  if (e >= Ef) return;
  int i0, i1; edge_nodes(e, E, src, dst, i0, i1);
  float s = a_node[i0] + b_node[i1] + gatb[0];
  s = s > 0.0f ? s : 0.2f * s;               // leaky_relu(0.2)
  s_e[e] = s;
  atomicMaxF(&smax[i0], s);
}

__global__ void asap_score2_kernel(const int* __restrict__ src, const int* __restrict__ dst,
                                   float* s_e, const float* __restrict__ smax,
                                   float* den, int E, int N) {
  int e = blockIdx.x * blockDim.x + threadIdx.x;
  int Ef = E + N;
  if (e >= Ef) return;
  int i0, i1; edge_nodes(e, E, src, dst, i0, i1);
  float ex = __expf(s_e[e] - smax[i0]);
  s_e[e] = ex;
  unsafeAtomicAdd(&den[i0], ex);
}

__global__ void asap_score3_kernel(const int* __restrict__ src, const int* __restrict__ dst,
                                   const float* __restrict__ s_e, const float* __restrict__ den,
                                   const float* __restrict__ x, float* outm, int E, int N) {
  int gid = blockIdx.x * blockDim.x + threadIdx.x;
  int e = gid >> 5, lane = gid & 31;
  int Ef = E + N;
  if (e >= Ef) return;
  int i0, i1; edge_nodes(e, E, src, dst, i0, i1);
  float score = s_e[e] / den[i0];
  #pragma unroll
  for (int j = 0; j < 3; ++j) {
    int c = lane + 32 * j;
    unsafeAtomicAdd(&outm[(size_t)i0 * C + c], score * x[(size_t)i1 * C + c]);
  }
}

// ---------------------------------------------------------------------------
// LEConv pieces
// ---------------------------------------------------------------------------
__global__ void asap_hle_kernel(const float* __restrict__ outm,
                                const float* __restrict__ lew,
                                const float* __restrict__ l1w, const float* __restrict__ l1b,
                                const float* __restrict__ l2w, const float* __restrict__ l2b,
                                float* hle, float* t1, float* t2, int N) {
  int n = blockIdx.x * blockDim.x + threadIdx.x;
  if (n >= N) return;
  float h = 0.0f, a1 = 0.0f, a2 = 0.0f;
  #pragma unroll 8
  for (int c = 0; c < C; ++c) {
    float v = outm[(size_t)n * C + c];
    h  += v * lew[c];
    a1 += v * l1w[c];
    a2 += v * l2w[c];
  }
  hle[n] = h;
  t1[n]  = a1 + l1b[0];
  t2[n]  = a2 + l2b[0];
}

__global__ void asap_aggr_kernel(const int* __restrict__ src, const int* __restrict__ dst,
                                 const float* __restrict__ hle, float* aggr, int E) {
  int e = blockIdx.x * blockDim.x + threadIdx.x;
  if (e < E) unsafeAtomicAdd(&aggr[src[e]], hle[dst[e]]);
}

__global__ void asap_final_kernel(const int* __restrict__ cnt0,
                                  const float* __restrict__ t1, const float* __restrict__ t2,
                                  const float* __restrict__ aggr, float* out, int N) {
  int n = blockIdx.x * blockDim.x + threadIdx.x;
  if (n >= N) return;
  float deg = (float)(cnt0[n] - 1);
  float res = deg * t1[n] + aggr[n] + t2[n];
  out[n] = sigmoidf_(res);
}

// ---------------------------------------------------------------------------
// host launcher
// ---------------------------------------------------------------------------
extern "C" void kernel_launch(void* const* d_in, const int* in_sizes, int n_in,
                              void* d_out, int out_size, void* d_ws, size_t ws_size,
                              hipStream_t stream) {
  const int N  = in_sizes[0] / C;
  const int E  = in_sizes[2] / 2;
  const int Ef = E + N;

  const float* x    = (const float*)d_in[0];
  const float* pos  = (const float*)d_in[1];
  const int*   ei   = (const int*)d_in[2];
  const int*   src  = ei;
  const int*   dst  = ei + E;
  const float* mw1  = (const float*)d_in[3];
  const float* mb1  = (const float*)d_in[4];
  const float* mw2  = (const float*)d_in[5];
  const float* mb2  = (const float*)d_in[6];
  const float* attw = (const float*)d_in[7];
  const float* attb = (const float*)d_in[8];
  // d_in[9..11] (pos_w1/pos_b1/pos_w2) and d_in[18] (coors_scale): dead code
  const float* uw1  = (const float*)d_in[12];
  const float* ub1  = (const float*)d_in[13];
  const float* uw2  = (const float*)d_in[14];
  const float* ub2  = (const float*)d_in[15];
  const float* lnw  = (const float*)d_in[16];
  const float* lnb  = (const float*)d_in[17];
  const float* lqw  = (const float*)d_in[19];
  const float* lqb  = (const float*)d_in[20];
  const float* gatw = (const float*)d_in[21];
  const float* gatb = (const float*)d_in[22];
  const float* lew  = (const float*)d_in[23];
  const float* l1w  = (const float*)d_in[24];
  const float* l1b  = (const float*)d_in[25];
  const float* l2w  = (const float*)d_in[26];
  const float* l2b  = (const float*)d_in[27];

  // -------- workspace carving (256B aligned) --------
  char* cur = (char*)d_ws;
  auto take = [&](size_t bytes) -> void* {
    void* p = (void*)cur;
    cur += (bytes + 255) & ~(size_t)255;
    return p;
  };
  int*    cnt0   = (int*)take((size_t)N * 4);
  int*    cnt1   = (int*)take((size_t)N * 4);
  float*  sAcc   = (float*)take(32);
  float*  sFin   = (float*)take(32);
  __bf16* w1T    = (__bf16*)take((size_t)C * 192 * 2);
  float*  w1last = (float*)take((size_t)C * 4);
  __bf16* w2T    = (__bf16*)take((size_t)C * C * 2);
  __bf16* u1T    = (__bf16*)take((size_t)C * 192 * 2);
  __bf16* u2T    = (__bf16*)take((size_t)C * C * 2);
  __bf16* lqT    = (__bf16*)take((size_t)C * C * 2);
  float*  agg    = (float*)take((size_t)N * C * 4);  // later reused as `outm`
  float*  x_pool = (float*)take((size_t)N * C * 4);
  float*  Xq     = (float*)take((size_t)N * C * 4);
  float*  Mq     = (float*)take((size_t)N * C * 4);
  float*  a_node = (float*)take((size_t)N * 4);
  float*  b_node = (float*)take((size_t)N * 4);
  float*  s_e    = (float*)take((size_t)Ef * 4);
  float*  smax   = (float*)take((size_t)N * 4);
  float*  den    = (float*)take((size_t)N * 4);
  float*  hle    = (float*)take((size_t)N * 4);
  float*  t1     = (float*)take((size_t)N * 4);
  float*  t2     = (float*)take((size_t)N * 4);
  float*  aggr   = (float*)take((size_t)N * 4);
  float*  outm   = agg;  // agg's last read is node_update; reuse for `out`

  const int NC = N * C;
  auto cdiv = [](long long a, long long b) { return (int)((a + b - 1) / b); };

  asap_init_kernel<<<cdiv(NC, 256), 256, 0, stream>>>(cnt0, cnt1, agg, Xq, smax, den, aggr, sAcc, N);
  asap_count_kernel<<<cdiv(E, 256), 256, 0, stream>>>(src, dst, cnt0, cnt1, E);
  asap_prep_weights_kernel<<<cdiv(C * 192, 256), 256, 0, stream>>>(mw1, mw2, uw1, uw2, lqw,
                                                                   w1T, w1last, w2T, u1T, u2T, lqT);
  asap_moments_kernel<<<cdiv(N, 256), 256, 0, stream>>>(x, cnt0, cnt1, sAcc, N);
  asap_finalize_stats_kernel<<<1, 32, 0, stream>>>(sAcc, sFin, Ef);

  asap_edge_mlp_kernel<<<2048, 64, 0, stream>>>(x, pos, src, dst, sFin, lnw, lnb,
                                                w1T, w1last, mb1, w2T, mb2,
                                                attw, attb, agg, E, N);
  asap_node_update_kernel<<<1024, 64, 0, stream>>>(x, agg, u1T, ub1, u2T, ub2, x_pool, N);

  asap_zero_kernel<<<cdiv(NC, 256), 256, 0, stream>>>(outm, NC);  // agg -> out buffer

  asap_xqmax_kernel<<<cdiv((long long)Ef * 32, 256), 256, 0, stream>>>(x_pool, src, dst, Xq, E, N);
  asap_mq_kernel<<<512, 64, 0, stream>>>(Xq, lqT, lqb, Mq, N);
  asap_node_dots_kernel<<<cdiv(N, 256), 256, 0, stream>>>(Mq, x_pool, gatw, a_node, b_node, N);

  asap_score1_kernel<<<cdiv(Ef, 256), 256, 0, stream>>>(src, dst, a_node, b_node, gatb, s_e, smax, E, N);
  asap_score2_kernel<<<cdiv(Ef, 256), 256, 0, stream>>>(src, dst, s_e, smax, den, E, N);
  asap_score3_kernel<<<cdiv((long long)Ef * 32, 256), 256, 0, stream>>>(src, dst, s_e, den, x, outm, E, N);

  asap_hle_kernel<<<cdiv(N, 256), 256, 0, stream>>>(outm, lew, l1w, l1b, l2w, l2b, hle, t1, t2, N);
  asap_aggr_kernel<<<cdiv(E, 256), 256, 0, stream>>>(src, dst, hle, aggr, E);
  asap_final_kernel<<<cdiv(N, 256), 256, 0, stream>>>(cnt0, t1, t2, aggr, (float*)d_out, N);
}
